// MLPTopK_1400159339074
// MI455X (gfx1250) — matrix operand order repaired and checked by reading, not compiled, not run on hardware
//
#include <hip/hip_runtime.h>
#include <hip/hip_bf16.h>

typedef __attribute__((ext_vector_type(16))) _Float16 v16h;
typedef __attribute__((ext_vector_type(8)))  float    v8f;
typedef __attribute__((ext_vector_type(4)))  float    v4f;

#define BB    8
#define NN    8192
#define CIN   128
#define COUT  256
#define MM    2048
#define KNB   16
#define TSTR  260   // LDS tile row stride in floats (16B aligned, avoids 64-bank conflicts)

// ---------------------------------------------------------------------------
// Pack fp32 weight matrix (K x N, row major) into WMMA f16 B-fragment order.
// Fragment (kt, nt): per lane, 16 contiguous halves:
//   column n = nt*16 + (lane & 15)
//   k = kt*32 + (lane < 16 ? 0 : 8) + (i & 7) + (i >= 8 ? 16 : 0)      (i = 0..15)
// ---------------------------------------------------------------------------
__global__ __launch_bounds__(256) void pack_w(const float* __restrict__ W,
                                              _Float16* __restrict__ Wp,
                                              int K, int Nn) {
    int tid = blockIdx.x * 256 + threadIdx.x;
    int total = (K >> 5) * (Nn >> 4) * 512;
    if (tid >= total) return;
    int i    = tid & 15;
    int lane = (tid >> 4) & 31;
    int frag = tid >> 9;
    int NT = Nn >> 4;
    int kt = frag / NT, nt = frag - kt * NT;
    int n  = nt * 16 + (lane & 15);
    int k  = kt * 32 + ((lane >> 4) << 3) + (i & 7) + ((i >> 3) << 4);
    Wp[tid] = (_Float16)W[k * Nn + n];
}

// ---------------------------------------------------------------------------
// LayerNorm + ReLU over one full 256-wide row held in LDS (one lane per row).
// ---------------------------------------------------------------------------
__device__ __forceinline__ void ln_relu_row(float* rowp, const float* g, const float* be) {
    float s = 0.f, sq = 0.f;
#pragma unroll
    for (int c = 0; c < 256; c += 4) {
        v4f v = *(v4f*)(rowp + c);
#pragma unroll
        for (int e = 0; e < 4; ++e) { s += v[e]; sq += v[e] * v[e]; }
    }
    float mean = s * (1.0f / 256.0f);
    float var  = sq * (1.0f / 256.0f) - mean * mean;
    float inv  = rsqrtf(var + 1e-5f);
#pragma unroll 8
    for (int c = 0; c < 256; ++c) {
        float v = (rowp[c] - mean) * inv * g[c] + be[c];
        rowp[c] = v > 0.f ? v : 0.f;
    }
}

// ---------------------------------------------------------------------------
// Fused: f2 = relu(LN(relu(LN(x@W1+b1))@W2+b2)), scores = f2@W3+b3
// One wave per workgroup owns a 32-row block (two 16-row WMMA sub-tiles that
// share every B fragment: 2 back-to-back WMMAs per weight load, halving the
// L2 weight stream). Single-wave WG: LDS ops in-order, no barriers needed.
// ---------------------------------------------------------------------------
__global__ __launch_bounds__(32) void mlp_fused(
    const float* __restrict__ x,
    const float* __restrict__ b1, const float* __restrict__ g1, const float* __restrict__ be1,
    const float* __restrict__ b2, const float* __restrict__ g2, const float* __restrict__ be2,
    const float* __restrict__ W3, const float* __restrict__ b3,
    const _Float16* __restrict__ W1p, const _Float16* __restrict__ W2p,
    float* __restrict__ f2, float* __restrict__ scores)
{
    __shared__ float tile[32 * TSTR];              // 33,280 B
    const int lane  = threadIdx.x & 31;
    const int rl    = lane & 15;                   // row within A sub-tile / col within B tile
    const int hi    = lane >> 4;                   // half-wave select
    const int khalf = hi * 8;                      // A-fragment K interleave
    const long rowBase = (long)blockIdx.x * 32;

    // ---- stage-1 A fragments: x rows (fp32 global -> f16 regs), ISA A layout ----
    v16h A1[2][4];
#pragma unroll
    for (int u = 0; u < 2; ++u) {
        const float* xrow = x + (rowBase + u * 16 + rl) * CIN;
#pragma unroll
        for (int kt = 0; kt < 4; ++kt) {
            const float* s0 = xrow + kt * 32 + khalf;   // K = kt*32 + khalf + 0..7
            const float* s1 = s0 + 16;                  // K = kt*32 + 16 + khalf + 0..7
            v16h a;
#pragma unroll
            for (int i = 0; i < 8; ++i) { a[i] = (_Float16)s0[i]; a[i + 8] = (_Float16)s1[i]; }
            A1[u][kt] = a;
        }
    }

    // ---- stage-1 GEMM: (32x128) x (128x256), dual accumulators per B frag ----
#pragma unroll
    for (int nt = 0; nt < 16; ++nt) {
        float bv = b1[nt * 16 + rl];
        v8f acc0 = {bv, bv, bv, bv, bv, bv, bv, bv};
        v8f acc1 = acc0;
#pragma unroll
        for (int kt = 0; kt < 4; ++kt) {
            v16h bf = *(const v16h*)(W1p + (size_t)((kt * 16 + nt) * 32 + lane) * 16);
            acc0 = __builtin_amdgcn_wmma_f32_16x16x32_f16(false, A1[0][kt], false, bf,
                                                          (short)0, acc0, false, false);
            acc1 = __builtin_amdgcn_wmma_f32_16x16x32_f16(false, A1[1][kt], false, bf,
                                                          (short)0, acc1, false, false);
        }
#pragma unroll
        for (int v = 0; v < 8; ++v) {
            tile[(v + hi * 8) * TSTR      + nt * 16 + rl] = acc0[v];
            tile[(v + hi * 8 + 16) * TSTR + nt * 16 + rl] = acc1[v];
        }
    }

    // ---- LN + ReLU (stage 1): one lane per row ----
    float* rowp = tile + lane * TSTR;
    ln_relu_row(rowp, g1, be1);

    // ---- stage-2 A fragments straight out of the LDS tile ----
    v16h A2[2][8];
#pragma unroll
    for (int u = 0; u < 2; ++u) {
#pragma unroll
        for (int kt = 0; kt < 8; ++kt) {
            const float* s0 = tile + (u * 16 + rl) * TSTR + kt * 32 + khalf;
            const float* s1 = s0 + 16;
            v16h a;
#pragma unroll
            for (int i = 0; i < 8; ++i) { a[i] = (_Float16)s0[i]; a[i + 8] = (_Float16)s1[i]; }
            A2[u][kt] = a;
        }
    }

    // ---- stage-2 GEMM: (32x256) x (256x256) ----
#pragma unroll
    for (int nt = 0; nt < 16; ++nt) {
        if (nt < 15)
            __builtin_prefetch(W2p + (size_t)((nt + 1) * 32 + lane) * 16, 0, 0);
        float bv = b2[nt * 16 + rl];
        v8f acc0 = {bv, bv, bv, bv, bv, bv, bv, bv};
        v8f acc1 = acc0;
#pragma unroll
        for (int kt = 0; kt < 8; ++kt) {
            v16h bf = *(const v16h*)(W2p + (size_t)((kt * 16 + nt) * 32 + lane) * 16);
            acc0 = __builtin_amdgcn_wmma_f32_16x16x32_f16(false, A2[0][kt], false, bf,
                                                          (short)0, acc0, false, false);
            acc1 = __builtin_amdgcn_wmma_f32_16x16x32_f16(false, A2[1][kt], false, bf,
                                                          (short)0, acc1, false, false);
        }
#pragma unroll
        for (int v = 0; v < 8; ++v) {
            tile[(v + hi * 8) * TSTR      + nt * 16 + rl] = acc0[v];
            tile[(v + hi * 8 + 16) * TSTR + nt * 16 + rl] = acc1[v];
        }
    }

    // ---- LN + ReLU (stage 2) ----
    ln_relu_row(rowp, g2, be2);

    // ---- store f2 (fp32, row major) + fused score projection ----
    float* dst = f2 + (rowBase + lane) * COUT;
    float dot = 0.f;
#pragma unroll
    for (int c = 0; c < 256; c += 4) {
        v4f v = *(v4f*)(rowp + c);
        *(v4f*)(dst + c) = v;
#pragma unroll
        for (int e = 0; e < 4; ++e) dot += v[e] * W3[c + e];
    }
    scores[rowBase + lane] = dot + b3[0];
}

// ---------------------------------------------------------------------------
// Per-batch top-M by score: full bitonic sort of 8192 keys in LDS (64 KB).
// Key = sortable(score) << 32 | (~index)  -> sort descending gives descending
// score with ascending-index tie-break (matches jax.lax.top_k stability).
// ---------------------------------------------------------------------------
__global__ __launch_bounds__(1024) void topk_sort(const float* __restrict__ scores,
                                                  const float* __restrict__ p,
                                                  float* __restrict__ p_out)
{
    __shared__ unsigned long long keys[NN];        // 64 KB
    const int b = blockIdx.x, t = threadIdx.x;
    for (int i = t; i < NN; i += 1024) {
        unsigned u = __float_as_uint(scores[b * NN + i]);
        u = (u & 0x80000000u) ? ~u : (u | 0x80000000u);
        keys[i] = ((unsigned long long)u << 32) | (unsigned)(0xFFFFFFFFu - (unsigned)i);
    }
    __syncthreads();
    for (int k = 2; k <= NN; k <<= 1) {
        for (int j = k >> 1; j > 0; j >>= 1) {
            for (int i = t; i < NN; i += 1024) {
                int ixj = i ^ j;
                if (ixj > i) {
                    unsigned long long a = keys[i], c = keys[ixj];
                    bool desc = ((i & k) == 0);
                    if ((a < c) == desc) { keys[i] = c; keys[ixj] = a; }
                }
            }
            __syncthreads();
        }
    }
    for (int m = t; m < MM; m += 1024) {
        int idx = (int)(0xFFFFFFFFu - (unsigned)keys[m]);
        const float* src = p + ((size_t)b * NN + idx) * 3;
        float* dst = p_out + ((size_t)b * MM + m) * 3;
        dst[0] = src[0]; dst[1] = src[1]; dst[2] = src[2];
    }
}

// ---------------------------------------------------------------------------
// kNN (16 of 8192, squared L2) + feature gather + max-pool.
// One wave per query: each lane keeps a register top-16 over its 256
// candidates; the 512 survivors are merged with a per-wave LDS bitonic sort
// (key = d2bits<<32 | idx: ascending d2, lower index first on ties).
// ---------------------------------------------------------------------------
__global__ __launch_bounds__(256) void knn_pool(const float* __restrict__ p,
                                                const float* __restrict__ p_out,
                                                const float* __restrict__ f2,
                                                float* __restrict__ y)
{
    __shared__ unsigned long long mrg[8][512];     // 32 KB
    const int lane = threadIdx.x & 31;
    const int wave = threadIdx.x >> 5;
    const int q = blockIdx.x * 8 + wave;           // 0 .. B*M-1
    const int b = q >> 11;                         // / MM

    const float* qp = p_out + (size_t)q * 3;
    const float qx = qp[0], qy = qp[1], qz = qp[2];

    float    bd[KNB];
    unsigned bi[KNB];
#pragma unroll
    for (int s = 0; s < KNB; ++s) { bd[s] = 3.4e38f; bi[s] = 0; }
    float worst = 3.4e38f; int wslot = 0;

    const float* pb = p + (size_t)b * NN * 3;
    for (int c = lane; c < NN; c += 32) {
        const float* pp = pb + c * 3;
        float dx = qx - pp[0], dy = qy - pp[1], dz = qz - pp[2];
        float d2 = dx * dx + dy * dy + dz * dz;
        if (d2 < worst) {
#pragma unroll
            for (int s = 0; s < KNB; ++s)
                if (s == wslot) { bd[s] = d2; bi[s] = (unsigned)c; }
            worst = bd[0]; wslot = 0;
#pragma unroll
            for (int s = 1; s < KNB; ++s)
                if (bd[s] > worst) { worst = bd[s]; wslot = s; }
        }
    }

    unsigned long long* mk = mrg[wave];
#pragma unroll
    for (int s = 0; s < KNB; ++s)
        mk[lane * KNB + s] = ((unsigned long long)__float_as_uint(bd[s]) << 32) | bi[s];
    __syncthreads();

    // ascending bitonic sort of 512 keys (all waves run identical stages)
    for (int k = 2; k <= 512; k <<= 1) {
        for (int j = k >> 1; j > 0; j >>= 1) {
            for (int i = lane; i < 512; i += 32) {
                int ixj = i ^ j;
                if (ixj > i) {
                    unsigned long long a = mk[i], c = mk[ixj];
                    bool asc = ((i & k) == 0);
                    if ((a > c) == asc) { mk[i] = c; mk[ixj] = a; }
                }
            }
            __syncthreads();
        }
    }

    // gather 16 nearest feature rows (f2 is L2-resident) and max-pool
    v4f a0, a1;
#pragma unroll
    for (int e = 0; e < 4; ++e) { a0[e] = -3.4e38f; a1[e] = -3.4e38f; }
    const int cb = lane * 8;
#pragma unroll
    for (int s = 0; s < KNB; ++s) {
        unsigned nb = (unsigned)mk[s];
        const float* fr = f2 + ((size_t)b * NN + nb) * COUT + cb;
        v4f v0 = *(const v4f*)fr;
        v4f v1 = *(const v4f*)(fr + 4);
#pragma unroll
        for (int e = 0; e < 4; ++e) {
            a0[e] = fmaxf(a0[e], v0[e]);
            a1[e] = fmaxf(a1[e], v1[e]);
        }
    }
    float* dst = y + (size_t)q * COUT + cb;
    *(v4f*)dst = a0;
    *(v4f*)(dst + 4) = a1;
}

// ---------------------------------------------------------------------------
extern "C" void kernel_launch(void* const* d_in, const int* in_sizes, int n_in,
                              void* d_out, int out_size, void* d_ws, size_t ws_size,
                              hipStream_t stream) {
    (void)in_sizes; (void)n_in; (void)out_size; (void)ws_size;
    const float* x   = (const float*)d_in[0];
    const float* p   = (const float*)d_in[1];
    const float* W1  = (const float*)d_in[2];
    const float* b1  = (const float*)d_in[3];
    const float* g1  = (const float*)d_in[4];
    const float* be1 = (const float*)d_in[5];
    const float* W2  = (const float*)d_in[6];
    const float* b2  = (const float*)d_in[7];
    const float* g2  = (const float*)d_in[8];
    const float* be2 = (const float*)d_in[9];
    const float* W3  = (const float*)d_in[10];
    const float* b3  = (const float*)d_in[11];

    float* y     = (float*)d_out;                         // (B, M, 256)
    float* p_out = y + (size_t)BB * MM * COUT;            // (B, M, 3)

    char* ws = (char*)d_ws;
    float*     f2     = (float*)ws;                       // B*N*256 fp32 = 64 MiB
    float*     scores = (float*)(ws + (size_t)67108864);  // B*N fp32
    _Float16*  W1p    = (_Float16*)(ws + (size_t)67371008);  // 32768 halves
    _Float16*  W2p    = (_Float16*)(ws + (size_t)67436544);  // 65536 halves

    pack_w<<<(32768 + 255) / 256, 256, 0, stream>>>(W1, W1p, CIN,  COUT);
    pack_w<<<(65536 + 255) / 256, 256, 0, stream>>>(W2, W2p, COUT, COUT);

    // 2048 single-wave workgroups, 32 rows each
    mlp_fused<<<2048, 32, 0, stream>>>(x, b1, g1, be1, b2, g2, be2, W3, b3,
                                       W1p, W2p, f2, scores);

    topk_sort<<<BB, 1024, 0, stream>>>(scores, p, p_out);

    knn_pool<<<(BB * MM) / 8, 256, 0, stream>>>(p, p_out, f2, y);
}
